// RGCNDistMult_14010183320204
// MI455X (gfx1250) — matrix operand order repaired and checked by reading, not compiled, not run on hardware
//
#include <hip/hip_runtime.h>

// ---------------------------------------------------------------------------
// RGCN (basis-decomposed) forward, 2 layers, CDNA5 / gfx1250, wave32 + WMMA.
//   msg_e = x[src_e] @ W_{etype_e},  W_r = sum_b comp[r,b] * basis[b]
//   agg[dst] += msg_e / cnt(dst, etype);  out = relu(agg + x@root + bias)
// Strategy: counting-sort edges by relation; one block = 32 same-relation
// edges as a 32x128 @ 128x128 bf16 WMMA GEMM (8 waves x 2 M-tiles x 4 K-steps
// of v_wmma_f32_16x16x32_bf16, B-fragment reused across M-tiles); scatter via
// branch-free global f32 atomics (padding rows carry norm=0, zeroed A rows).
// ---------------------------------------------------------------------------

typedef __attribute__((ext_vector_type(16))) __bf16 v16bf;
typedef __attribute__((ext_vector_type(8)))  float  v8f;

union ABfrag { uint4 q[2]; v16bf v; };

__device__ __forceinline__ unsigned short f2bf(float f) {
    unsigned u = __float_as_uint(f);
    u += 0x7FFFu + ((u >> 16) & 1u);          // round-to-nearest-even
    return (unsigned short)(u >> 16);
}

// ------------------------------ utility kernels ----------------------------

__global__ void k_zero_u32(unsigned* __restrict__ p, long n) {
    long i = (long)blockIdx.x * blockDim.x + threadIdx.x;
    long s = (long)gridDim.x * blockDim.x;
    for (; i < n; i += s) p[i] = 0u;
}

__global__ void k_zero_f32(float* __restrict__ p, long n) {
    long i = (long)blockIdx.x * blockDim.x + threadIdx.x;
    long s = (long)gridDim.x * blockDim.x;
    for (; i < n; i += s) p[i] = 0.0f;
}

__global__ void k_cvt_bf16(const float* __restrict__ x, unsigned short* __restrict__ o, long n) {
    long i = (long)blockIdx.x * blockDim.x + threadIdx.x;
    long s = (long)gridDim.x * blockDim.x;
    for (; i < n; i += s) o[i] = f2bf(x[i]);
}

// --------------------- per-(dst,relation) normalization --------------------

__global__ void k_count(const long long* __restrict__ dst, const long long* __restrict__ et,
                        unsigned* __restrict__ cnt, long E, int R) {
    long e = (long)blockIdx.x * blockDim.x + threadIdx.x;
    if (e < E) atomicAdd(&cnt[(size_t)dst[e] * R + (size_t)et[e]], 1u);
}

__global__ void k_norm(const long long* __restrict__ dst, const long long* __restrict__ et,
                       const unsigned* __restrict__ cnt, float* __restrict__ nrm, long E, int R) {
    long e = (long)blockIdx.x * blockDim.x + threadIdx.x;
    if (e < E) {
        unsigned c = cnt[(size_t)dst[e] * R + (size_t)et[e]];
        nrm[e] = 1.0f / (float)(c > 1u ? c : 1u);
    }
}

// ----------------------- counting sort of edges by relation ----------------

__global__ void k_hist(const long long* __restrict__ et, unsigned* __restrict__ hist, long E) {
    long e = (long)blockIdx.x * blockDim.x + threadIdx.x;
    if (e < E) atomicAdd(&hist[(int)et[e]], 1u);
}

__global__ void k_prefix(const unsigned* __restrict__ hist, unsigned* __restrict__ bstart,
                         unsigned* __restrict__ cursor, unsigned* __restrict__ tstart,
                         unsigned* __restrict__ meta, int R) {
    if (blockIdx.x == 0 && threadIdx.x == 0) {
        unsigned b = 0, t = 0;
        for (int r = 0; r < R; ++r) {
            bstart[r] = b; cursor[r] = b; tstart[r] = t;
            unsigned h = hist[r];
            b += h; t += (h + 31u) >> 5;      // 32-edge tiles
        }
        bstart[R] = b; tstart[R] = t; meta[0] = t;
    }
}

__global__ void k_scatter(const long long* __restrict__ et, unsigned* __restrict__ cursor,
                          int* __restrict__ sorted, long E) {
    long e = (long)blockIdx.x * blockDim.x + threadIdx.x;
    if (e < E) {
        unsigned p = atomicAdd(&cursor[(int)et[e]], 1u);
        sorted[p] = (int)e;
    }
}

// ------------- W^T[r][n][k] = bf16( sum_b comp[r,b]*basis[b][k][n] ) -------
// LDS-tiled 32x32 so basis reads and Wt writes are both coalesced; makes the
// WMMA B-fragment (contiguous K) loadable as two global_load_b128 per step.

__global__ __launch_bounds__(256) void k_wbuild(const float* __restrict__ comp,
                                                const float* __restrict__ basis,
                                                unsigned short* __restrict__ Wt,
                                                int Bn) {
    __shared__ float bt[32][33];
    const int D = 128;
    int r  = blockIdx.z;
    int n0 = blockIdx.y * 32;
    int k0 = blockIdx.x * 32;
    int tid = threadIdx.x;
    float acc[4] = {0.f, 0.f, 0.f, 0.f};
    for (int b = 0; b < Bn; ++b) {
        float cb = comp[(size_t)r * Bn + b];
        __syncthreads();
#pragma unroll
        for (int i = 0; i < 4; ++i) {
            int lin = tid + i * 256;
            int kk = lin >> 5, nn = lin & 31;
            bt[kk][nn] = basis[((size_t)b * D + (k0 + kk)) * D + (n0 + nn)];
        }
        __syncthreads();
#pragma unroll
        for (int i = 0; i < 4; ++i) {
            int lin = tid + i * 256;
            int nl = lin >> 5, kl = lin & 31;
            acc[i] += cb * bt[kl][nl];
        }
    }
#pragma unroll
    for (int i = 0; i < 4; ++i) {
        int lin = tid + i * 256;
        int nl = lin >> 5, kl = lin & 31;
        Wt[((size_t)r * D + (n0 + nl)) * D + (k0 + kl)] = f2bf(acc[i]);
    }
}

__global__ void k_roott(const float* __restrict__ root, unsigned short* __restrict__ rt) {
    const int D = 128;
    int idx = blockIdx.x * 256 + threadIdx.x;   // 64 blocks x 256 = 16384
    int n = idx >> 7, k = idx & 127;
    rt[(size_t)n * D + k] = f2bf(root[(size_t)k * D + n]);
}

// --------------------------- message WMMA kernel ---------------------------
// One block = 32 same-relation edges. 8 waves = 8 column tiles of D=128.
// Each wave: 2 M-tiles x 4 K-steps of v_wmma_f32_16x16x32_bf16 with the
// B-fragment reused across M-tiles; branch-free norm-scaled f32 atomic
// scatter (padding rows: zeroed A, norm = 0, dst offset 0).

__global__ __launch_bounds__(256) void k_msg(const unsigned short* __restrict__ xb,
                                             const unsigned short* __restrict__ Wt,
                                             const int* __restrict__ sorted,
                                             const unsigned* __restrict__ tstart,
                                             const unsigned* __restrict__ bstart,
                                             const unsigned* __restrict__ hist,
                                             const unsigned* __restrict__ meta,
                                             const long long* __restrict__ src,
                                             const long long* __restrict__ dst,
                                             const float* __restrict__ nrm,
                                             float* __restrict__ agg, int R) {
    const int D = 128;
    __shared__ uint4 xt4[32][17];                     // 32 rows x 128 bf16 (+16B pad)
    __shared__ int   s_src[32];
    __shared__ __align__(16) int   s_doff[32];        // dst*D as int offset
    __shared__ __align__(16) float s_nrm[32];         // 0 for padding rows
    __shared__ int   s_info[3];                       // r, base, nvalid

    int t = blockIdx.x;
    if (t >= (int)meta[0]) return;                    // uniform per block
    int tid = threadIdx.x;

    if (tid == 0) {
        int lo = 0, hi = R;                           // tstart[lo] <= t < tstart[lo+1]
        while (hi - lo > 1) { int mid = (lo + hi) >> 1; if ((int)tstart[mid] <= t) lo = mid; else hi = mid; }
        int r = lo;
        int tib  = t - (int)tstart[r];
        int base = (int)bstart[r] + tib * 32;
        int rem  = (int)hist[r] - tib * 32;
        s_info[0] = r; s_info[1] = base; s_info[2] = rem > 32 ? 32 : rem;
    }
    __syncthreads();
    int r = s_info[0], base = s_info[1], nv = s_info[2];

    if (tid < 32) {
        if (tid < nv) {
            int e = sorted[base + tid];
            s_src[tid]  = (int)src[e];
            s_doff[tid] = (int)dst[e] * D;
            s_nrm[tid]  = nrm[e];
        } else {
            s_src[tid] = 0; s_doff[tid] = 0; s_nrm[tid] = 0.0f;
        }
    }
    __syncthreads();

    // gather A tile: 32 rows x 128 bf16; each thread moves 2x16B (uint4)
#pragma unroll
    for (int i = 0; i < 2; ++i) {
        int lin = tid + i * 256;
        int row = lin >> 4, seg = lin & 15;
        uint4 q = {0u, 0u, 0u, 0u};
        if (row < nv)
            q = *(const uint4*)(xb + (size_t)s_src[row] * D + seg * 8);
        xt4[row][seg] = q;
    }
    __syncthreads();

    int lane = tid & 31;
    int wv   = tid >> 5;                   // column tile 0..7
    int rowA = lane & 15;                  // A: M = lane&15 for both halves
    int half = lane >> 4;                  // 0: K 0-7/16-23, 1: K 8-15/24-31
    const unsigned short* wcol = Wt + ((size_t)r * D + (wv * 16 + (lane & 15))) * D;

    v8f acc0 = {};                         // edges 0..15
    v8f acc1 = {};                         // edges 16..31
#pragma unroll
    for (int c = 0; c < 4; ++c) {
        ABfrag a0, a1, b;
        a0.q[0] = xt4[rowA][c * 4 + half];             // K = c*32 + half*8 + [0..7]
        a0.q[1] = xt4[rowA][c * 4 + half + 2];         // K = +16
        a1.q[0] = xt4[rowA + 16][c * 4 + half];
        a1.q[1] = xt4[rowA + 16][c * 4 + half + 2];
        const uint4* wq = (const uint4*)(wcol + c * 32 + half * 8);
        b.q[0] = wq[0];                                // K = c*32 + half*8 + [0..7]
        b.q[1] = wq[2];                                // K = +16
        acc0 = __builtin_amdgcn_wmma_f32_16x16x32_bf16(false, a0.v, false, b.v,
                                                       (short)0, acc0, false, false);
        acc1 = __builtin_amdgcn_wmma_f32_16x16x32_bf16(false, a1.v, false, b.v,
                                                       (short)0, acc1, false, false);
    }

    // branch-free scatter: lanes 0-15 -> M=v, lanes 16-31 -> M=v+8
    int nn   = wv * 16 + (lane & 15);
    int moff = (lane < 16) ? 0 : 8;
    uint4  o0 = *(const uint4*)&s_doff[moff];
    uint4  o1 = *(const uint4*)&s_doff[moff + 4];
    uint4  o2 = *(const uint4*)&s_doff[16 + moff];
    uint4  o3 = *(const uint4*)&s_doff[16 + moff + 4];
    float4 n0 = *(const float4*)&s_nrm[moff];
    float4 n1 = *(const float4*)&s_nrm[moff + 4];
    float4 n2 = *(const float4*)&s_nrm[16 + moff];
    float4 n3 = *(const float4*)&s_nrm[16 + moff + 4];
    int   od[16] = {(int)o0.x,(int)o0.y,(int)o0.z,(int)o0.w,
                    (int)o1.x,(int)o1.y,(int)o1.z,(int)o1.w,
                    (int)o2.x,(int)o2.y,(int)o2.z,(int)o2.w,
                    (int)o3.x,(int)o3.y,(int)o3.z,(int)o3.w};
    float nd[16] = {n0.x,n0.y,n0.z,n0.w, n1.x,n1.y,n1.z,n1.w,
                    n2.x,n2.y,n2.z,n2.w, n3.x,n3.y,n3.z,n3.w};
    float vd[16] = {acc0[0],acc0[1],acc0[2],acc0[3],acc0[4],acc0[5],acc0[6],acc0[7],
                    acc1[0],acc1[1],acc1[2],acc1[3],acc1[4],acc1[5],acc1[6],acc1[7]};
#pragma unroll
    for (int v = 0; v < 16; ++v)
        atomicAdd(&agg[(size_t)(od[v] + nn)], vd[v] * nd[v]);
}

// ------------------ self-loop: out = relu(agg + x@root + bias) -------------

__global__ __launch_bounds__(256) void k_self(const unsigned short* __restrict__ xb,
                                              const unsigned short* __restrict__ rt,
                                              const float* __restrict__ bias,
                                              float* __restrict__ xo, int nRows) {
    const int D = 128;
    int tile = blockIdx.x;
    int tid = threadIdx.x, lane = tid & 31, wv = tid >> 5;
    int rowA = lane & 15;
    int half = lane >> 4;
    int gRow = tile * 16 + rowA;

    const unsigned short* xrow = xb + (size_t)gRow * D;
    const unsigned short* rcol = rt + (size_t)(wv * 16 + (lane & 15)) * D;

    v8f acc = {};
#pragma unroll
    for (int c = 0; c < 4; ++c) {
        ABfrag a, b;
        if (gRow < nRows) {
            a.q[0] = *(const uint4*)(xrow + c * 32 + half * 8);
            a.q[1] = *(const uint4*)(xrow + c * 32 + half * 8 + 16);
        } else {
            uint4 z = {0u, 0u, 0u, 0u}; a.q[0] = z; a.q[1] = z;
        }
        b.q[0] = *(const uint4*)(rcol + c * 32 + half * 8);
        b.q[1] = *(const uint4*)(rcol + c * 32 + half * 8 + 16);
        acc = __builtin_amdgcn_wmma_f32_16x16x32_bf16(false, a.v, false, b.v,
                                                      (short)0, acc, false, false);
    }

    int col  = wv * 16 + (lane & 15);
    int moff = (lane < 16) ? 0 : 8;
    float bi = bias[col];
#pragma unroll
    for (int v = 0; v < 8; ++v) {
        int row = tile * 16 + v + moff;
        if (row < nRows) {
            size_t i = (size_t)row * D + col;
            float val = xo[i] + acc[v] + bi;
            xo[i] = val > 0.0f ? val : 0.0f;
        }
    }
}

// --------------------------------- launch ----------------------------------

extern "C" void kernel_launch(void* const* d_in, const int* in_sizes, int n_in,
                              void* d_out, int out_size, void* d_ws, size_t ws_size,
                              hipStream_t stream) {
    (void)n_in; (void)out_size; (void)ws_size;
    const float*     entity = (const float*)d_in[0];
    const float*     comp   = (const float*)d_in[1];   // [L,R,B]
    const float*     basis  = (const float*)d_in[2];   // [L,B,D,D]
    const float*     root   = (const float*)d_in[3];   // [L,D,D]
    const float*     bias   = (const float*)d_in[4];   // [L,D]
    const long long* eidx   = (const long long*)d_in[5]; // [2,E] int64
    const long long* etyp   = (const long long*)d_in[6]; // [E]   int64

    const int  D = 128, R = 200, Bn = 50, Lyr = 2;
    const long N = (long)in_sizes[0] / D;
    const long E = (long)in_sizes[6];
    const long long* srcIdx = eidx;
    const long long* dstIdx = eidx + E;

    // ---- carve workspace (cnt region reused for x_bf16 + x_mid) ----
    char* ws = (char*)d_ws;
    size_t off = 0;
    auto carve = [&](size_t bytes) { char* p = ws + off; off += (bytes + 255) & ~(size_t)255; return p; };

    size_t xb_bytes   = (size_t)N * D * 2;                 // 25.6 MB
    size_t xmid_bytes = (size_t)N * D * 4;                 // 51.2 MB
    size_t cnt_bytes  = (size_t)N * R * 4;                 // 80   MB
    size_t reuse = cnt_bytes;
    size_t need  = ((xb_bytes + 255) & ~(size_t)255) + xmid_bytes;
    if (need > reuse) reuse = need;
    char* region = carve(reuse);
    unsigned*       cnt  = (unsigned*)region;
    unsigned short* xb   = (unsigned short*)region;
    float*          xmid = (float*)(region + ((xb_bytes + 255) & ~(size_t)255));

    float*          nrm    = (float*)carve((size_t)E * 4);
    int*            sorted = (int*)carve((size_t)E * 4);
    unsigned short* Wt     = (unsigned short*)carve((size_t)R * D * D * 2);
    unsigned short* Rt     = (unsigned short*)carve((size_t)D * D * 2);
    unsigned*       hist   = (unsigned*)carve(1024);
    unsigned*       bstart = (unsigned*)carve(1024);
    unsigned*       cursor = (unsigned*)carve(1024);
    unsigned*       tstart = (unsigned*)carve(1024);
    unsigned*       meta   = (unsigned*)carve(256);

    int eb = (int)((E + 255) / 256);

    // ---- once: normalization counts + relation bucketing ----
    k_zero_u32<<<2048, 256, 0, stream>>>(cnt, N * (long)R);
    k_count  <<<eb, 256, 0, stream>>>(dstIdx, etyp, cnt, E, R);
    k_norm   <<<eb, 256, 0, stream>>>(dstIdx, etyp, cnt, nrm, E, R);
    k_zero_u32<<<1, 256, 0, stream>>>(hist, R);
    k_hist   <<<eb, 256, 0, stream>>>(etyp, hist, E);
    k_prefix <<<1, 32, 0, stream>>>(hist, bstart, cursor, tstart, meta, R);
    k_scatter<<<eb, 256, 0, stream>>>(etyp, cursor, sorted, E);

    // ---- layers ----
    const float* xin = entity;
    int maxTiles = (int)(E / 32 + R + 1);
    int rowTiles = (int)((N + 15) / 16);
    for (int l = 0; l < Lyr; ++l) {
        float* xo = (l == 0) ? xmid : (float*)d_out;
        k_cvt_bf16<<<4096, 256, 0, stream>>>(xin, xb, N * (long)D);
        dim3 gW(4, 4, R);
        k_wbuild<<<gW, 256, 0, stream>>>(comp + (size_t)l * R * Bn,
                                         basis + (size_t)l * Bn * D * D, Wt, Bn);
        k_roott<<<64, 256, 0, stream>>>(root + (size_t)l * D * D, Rt);
        k_zero_f32<<<4096, 256, 0, stream>>>(xo, N * (long)D);
        k_msg<<<maxTiles, 256, 0, stream>>>(xb, Wt, sorted, tstart, bstart, hist, meta,
                                            srcIdx, dstIdx, nrm, xo, R);
        k_self<<<rowTiles, 256, 0, stream>>>(xb, Rt, bias + (size_t)l * D, xo, (int)N);
        xin = xo;
    }
}
